// FrequencySparseAttention_81252191306588
// MI455X (gfx1250) — matrix operand order repaired and checked by reading, not compiled, not run on hardware
//
#include <hip/hip_runtime.h>
#include <hip/hip_bf16.h>

// ---------------------------------------------------------------------------
// FrequencySparseAttention for gfx1250 (MI455X), bf16 WMMA pipeline:
//   1) Q/K/V projections: bf16 WMMA GEMM (fp32 inputs converted while staging
//      to LDS), epilogue scatters to head-major layouts (V stored transposed).
//   2) Flash attention with energy-based key masking, fp32 online softmax,
//      WMMA for Q*K^T and P*V; K/V tiles staged by the Tensor Data Mover
//      (tensor_load_to_lds + s_wait_tensorcnt) with LDS row padding done by
//      the TDM pad fields.
//   3) Output projection: bf16 WMMA GEMM, fp32 result to d_out.
// ---------------------------------------------------------------------------

typedef __attribute__((ext_vector_type(16))) __bf16   v16bf;
typedef __attribute__((ext_vector_type(8)))  float    v8f;
typedef __attribute__((ext_vector_type(4)))  unsigned u32x4;
typedef __attribute__((ext_vector_type(8)))  unsigned u32x8;

#define WMMA_BF16(a, b, c) \
  __builtin_amdgcn_wmma_f32_16x16x32_bf16(false, (a), false, (b), (short)0, (c), false, false)

// A-matrix fragment (16x32 bf16). Lane = m + 16*h.
// Element j: k = 8h + j (j<8), k = 16 + 8h + (j-8) (j>=8).  Two 16B loads.
static __device__ __forceinline__ v16bf fragA(const __bf16* p, int h) {
  union { v16bf v; uint4 u[2]; } r;
  r.u[0] = *(const uint4*)(p + 8 * h);
  r.u[1] = *(const uint4*)(p + 16 + 8 * h);
  return r.v;
}

// B-matrix fragment (32x16 bf16) gathered from storage S[n][k] (row per n),
// i.e. B[k][n] = S[n][k]. Lane = n + 16*h. Element j: k = 8h + j (j<8), k = 16 + 8h + (j-8) (j>=8).
static __device__ __forceinline__ v16bf fragB(const __bf16* p, int h) {
  union { v16bf v; uint4 u[2]; } r;
  r.u[0] = *(const uint4*)(p + 8 * h);
  r.u[1] = *(const uint4*)(p + 16 + 8 * h);
  return r.v;
}

static __device__ __forceinline__ void cvt8_store(const float* src, __bf16* dst) {
  float4 a = *(const float4*)src;
  float4 b = *(const float4*)(src + 4);
  __bf16 t[8] = {(__bf16)a.x, (__bf16)a.y, (__bf16)a.z, (__bf16)a.w,
                 (__bf16)b.x, (__bf16)b.y, (__bf16)b.z, (__bf16)b.w};
  *(uint4*)dst = *(uint4*)t;
}

// ---------------------------------------------------------------------------
// Tensor Data Mover: 2D bf16 tile (tile_d0 x tile_d1 elements) from global
// (row stride = stride0 elements) into LDS at lds_addr, padding the LDS
// destination by pad_amount DWORDs after every pad_interval DWORDs.
// D# layout per CDNA5 ISA §8.3/§8.4; 2-group (2D) form of tensor_load_to_lds.
// Must be executed by a single wave (scalar operands, EXEC ignored).
// ---------------------------------------------------------------------------
static __device__ __forceinline__ void tdm_load_2d_bf16(
    unsigned lds_addr, const void* gptr, unsigned tile_d0, unsigned tile_d1,
    unsigned stride0, unsigned pad_interval_code, unsigned pad_amount_code) {
  const unsigned long long ga = (unsigned long long)(uintptr_t)gptr;
  u32x4 g0;
  g0[0] = 0x1u;                                           // count=1, user D#
  g0[1] = lds_addr;                                       // LDS byte address
  g0[2] = (unsigned)ga;                                   // global_addr[31:0]
  g0[3] = ((unsigned)(ga >> 32) & 0x01FFFFFFu) | (2u << 30);  // [56:32], type=2
  u32x8 g1;
  g1[0] = (1u << 16)                                      // data_size = 2 bytes
        | (1u << 20)                                      // pad_enable
        | (pad_interval_code << 22) | (pad_amount_code << 25);
  g1[1] = (tile_d0 & 0xFFFFu) << 16;                      // tensor_dim0[15:0]
  g1[2] = (tile_d0 >> 16) | ((tile_d1 & 0xFFFFu) << 16);  // dim0 hi | dim1 lo
  g1[3] = (tile_d1 >> 16) | (tile_d0 << 16);              // dim1 hi | tile_dim0
  g1[4] = tile_d1;                                        // tile_dim1 (dim2=0)
  g1[5] = stride0;                                        // tensor_dim0_stride lo
  g1[6] = 0u;                                             // stride0 hi | stride1 lo
  g1[7] = 0u;                                             // stride1 hi
  asm volatile("tensor_load_to_lds %0, %1" :: "s"(g0), "s"(g1) : "memory");
}

// ---------------------------------------------------------------------------
// Problem constants (reference: B=4, L=2048, D=1024, H=16, Dh=64, C_E=8).
// ---------------------------------------------------------------------------
constexpr int Dm   = 1024;   // model dim (GEMM N and K)
constexpr int Lseq = 2048;
constexpr int Bb   = 4;
constexpr int Hh   = 16;
constexpr int Dh   = 64;
constexpr int Mrows = Bb * Lseq;  // 8192

constexpr int BM = 128, BN = 64, BK = 32;

enum EpMode { EP_Q = 0, EP_K = 1, EP_V = 2, EP_OUT = 3 };

// ---------------------------------------------------------------------------
// GEMM: D = A(MxK) @ W(NxK)^T + bias.  256 threads = 8 waves, each wave 32x32.
// ---------------------------------------------------------------------------
template <bool ASrcF32, int EP>
__global__ __launch_bounds__(256) void gemm_kernel(const void* __restrict__ Ain,
                                                   const float* __restrict__ W,
                                                   const float* __restrict__ bias,
                                                   void* __restrict__ Dst) {
  __shared__ __bf16 As[BM][BK + 8];
  __shared__ __bf16 Bs[BN][BK + 8];

  const int tid = threadIdx.x;
  const int m0 = blockIdx.x * BM, n0 = blockIdx.y * BN;
  const int lane = tid & 31, wave = tid >> 5;
  const int wm = wave & 3, wn = wave >> 2;   // 4x2 wave grid of 32x32 tiles
  const int ml = lane & 15, hl = lane >> 4;

  const v8f zero8 = {0.f, 0.f, 0.f, 0.f, 0.f, 0.f, 0.f, 0.f};
  v8f acc[2][2];
#pragma unroll
  for (int i = 0; i < 2; i++)
#pragma unroll
    for (int j = 0; j < 2; j++) acc[i][j] = zero8;

  for (int k0 = 0; k0 < Dm; k0 += BK) {
    __syncthreads();
    // Stage A tile (128x32) -> bf16 LDS. 512 8-elem slots, 2 per thread.
#pragma unroll
    for (int i = 0; i < 2; i++) {
      int slot = tid + i * 256;
      int row = slot >> 2, kc = (slot & 3) * 8;
      if constexpr (ASrcF32) {
        const float* s = (const float*)Ain + (size_t)(m0 + row) * Dm + k0 + kc;
        cvt8_store(s, &As[row][kc]);
      } else {
        const __bf16* s = (const __bf16*)Ain + (size_t)(m0 + row) * Dm + k0 + kc;
        *(uint4*)&As[row][kc] = *(const uint4*)s;
      }
    }
    // Stage W tile (64 n-rows x 32 k) -> bf16 LDS. 256 slots, 1 per thread.
    {
      int row = tid >> 2, kc = (tid & 3) * 8;
      const float* s = W + (size_t)(n0 + row) * Dm + k0 + kc;
      cvt8_store(s, &Bs[row][kc]);
    }
    __syncthreads();

    v16bf af[2], bfv[2];
    af[0]  = fragA(&As[wm * 32 + ml][0], hl);
    af[1]  = fragA(&As[wm * 32 + 16 + ml][0], hl);
    bfv[0] = fragB(&Bs[wn * 32 + ml][0], hl);
    bfv[1] = fragB(&Bs[wn * 32 + 16 + ml][0], hl);
#pragma unroll
    for (int mt = 0; mt < 2; mt++)
#pragma unroll
      for (int nt = 0; nt < 2; nt++)
        acc[mt][nt] = WMMA_BF16(af[mt], bfv[nt], acc[mt][nt]);
  }

  // Epilogue. C-frag: row = r + 8*hl (+16*mt), col = ml (+16*nt).
#pragma unroll
  for (int nt = 0; nt < 2; nt++) {
    const int gcol = n0 + wn * 32 + nt * 16 + ml;
    const float bv = bias[gcol];
#pragma unroll
    for (int mt = 0; mt < 2; mt++) {
#pragma unroll
      for (int r = 0; r < 8; r++) {
        const int grow = m0 + wm * 32 + mt * 16 + r + 8 * hl;
        float val = acc[mt][nt][r] + bv;
        if constexpr (EP == EP_OUT) {
          ((float*)Dst)[(size_t)grow * Dm + gcol] = val;
        } else {
          const int bb = grow >> 11, lq = grow & (Lseq - 1);
          const int hh = gcol >> 6, dd = gcol & (Dh - 1);
          if constexpr (EP == EP_Q)  // fold 1/sqrt(Dh) into Q
            ((__bf16*)Dst)[((size_t)(bb * Hh + hh) * Lseq + lq) * Dh + dd] =
                (__bf16)(val * 0.125f);
          else if constexpr (EP == EP_K)
            ((__bf16*)Dst)[((size_t)(bb * Hh + hh) * Lseq + lq) * Dh + dd] = (__bf16)val;
          else  // EP_V: store transposed [B][H][Dh][L]
            ((__bf16*)Dst)[((size_t)(bb * Hh + hh) * Dh + dd) * Lseq + lq] = (__bf16)val;
        }
      }
    }
  }
}

// ---------------------------------------------------------------------------
// Flash attention with energy-based key mask. 128 threads = 4 waves; each
// wave owns 16 query rows of a 64-row query tile; 64-key blocks per step.
// K and V^T tiles are staged into LDS by the Tensor Data Mover.
// ---------------------------------------------------------------------------
__global__ __launch_bounds__(128) void attn_kernel(const __bf16* __restrict__ Qp,
                                                   const __bf16* __restrict__ Kp,
                                                   const __bf16* __restrict__ Vt,
                                                   const float* __restrict__ ec,
                                                   __bf16* __restrict__ Out) {
  __shared__ __bf16 Ks[64][72];       // keys x Dh   (72 = 64 + TDM pad of 4 DW)
  __shared__ __bf16 Vs[64][72];       // Dh x keys (from transposed V)
  __shared__ __bf16 Ps[4][16][72];    // per-wave P tile (16 q x 64 keys)
  __shared__ float  msk[64];

  const int tid = threadIdx.x, lane = tid & 31, wave = tid >> 5;
  const int ml = lane & 15, hl = lane >> 4;
  const int qt = blockIdx.x & 31, bh = blockIdx.x >> 5;
  const int b = bh >> 4, h = bh & 15;

  const __bf16* Qb = Qp + (size_t)bh * Lseq * Dh;
  const __bf16* Kb = Kp + (size_t)bh * Lseq * Dh;
  const __bf16* Vb = Vt + (size_t)bh * Dh * Lseq;
  const int q0 = qt * 64 + wave * 16;

  // LDS byte offsets (low 32 bits of a generic pointer to LDS = LDS address).
  const unsigned ks_lds = (unsigned)(uintptr_t)&Ks[0][0];
  const unsigned vs_lds = (unsigned)(uintptr_t)&Vs[0][0];

  // Q fragments for this wave's 16 rows (Dh=64 -> two K=32 fragments).
  v16bf qf[2];
  qf[0] = fragA(Qb + (size_t)(q0 + ml) * Dh, hl);
  qf[1] = fragA(Qb + (size_t)(q0 + ml) * Dh + 32, hl);

  const v8f zero8 = {0.f, 0.f, 0.f, 0.f, 0.f, 0.f, 0.f, 0.f};
  float mrow[8], lrow[8];
  v8f o[4];
#pragma unroll
  for (int r = 0; r < 8; r++) { mrow[r] = -3.0e38f; lrow[r] = 0.f; }
#pragma unroll
  for (int d = 0; d < 4; d++) o[d] = zero8;

  for (int kb = 0; kb < Lseq / 64; kb++) {
    __syncthreads();  // previous tiles fully consumed before TDM overwrites
    if (wave == 0) {
      // K tile: 64 keys x 64 bf16, contiguous rows (stride 64 elements).
      // Pad LDS rows: 32 DWORDs data (code 4) + 4 DWORDs pad (code 3) -> 72 elems.
      tdm_load_2d_bf16(ks_lds, Kb + (size_t)kb * 64 * Dh, 64, 64, 64, 4, 3);
      // V^T tile: 64 Dh-rows x 64 keys, row stride 2048 elements.
      tdm_load_2d_bf16(vs_lds, Vb + (size_t)kb * 64, 64, 64, (unsigned)Lseq, 4, 3);
      __builtin_amdgcn_s_wait_tensorcnt(0);
    }
    if (tid < 64) {  // energy mask: mean(|coeffs|) > 0.1
      const float* e = ec + ((size_t)b * Lseq + kb * 64 + tid) * 8;
      float s = 0.f;
#pragma unroll
      for (int j = 0; j < 8; j++) s += fabsf(e[j]);
      msk[tid] = (s * 0.125f > 0.1f) ? 1.f : 0.f;
    }
    if (lane == 0 && kb + 1 < Lseq / 64) {  // global_prefetch next K tile
      __builtin_prefetch(Kb + (size_t)(kb + 1) * 64 * Dh + wave * 1024, 0, 1);
    }
    __syncthreads();  // TDM data + mask visible to all waves

    // S = Q * K^T (Q pre-scaled by 1/sqrt(Dh)).
    v8f s4[4];
#pragma unroll
    for (int nt = 0; nt < 4; nt++) {
      v8f a = zero8;
#pragma unroll
      for (int ks = 0; ks < 2; ks++) {
        v16bf kf = fragB(&Ks[nt * 16 + ml][32 * ks], hl);
        a = WMMA_BF16(qf[ks], kf, a);
      }
      s4[nt] = a;
    }

    float keep[4];
#pragma unroll
    for (int nt = 0; nt < 4; nt++) keep[nt] = msk[nt * 16 + ml];

    // Row max over kept keys (columns live across 16 lanes per half-wave).
    float rmax[8];
#pragma unroll
    for (int r = 0; r < 8; r++) rmax[r] = -3.0e38f;
#pragma unroll
    for (int nt = 0; nt < 4; nt++)
#pragma unroll
      for (int r = 0; r < 8; r++) {
        float sv = (keep[nt] != 0.f) ? s4[nt][r] : -3.0e38f;
        rmax[r] = fmaxf(rmax[r], sv);
      }
#pragma unroll
    for (int d = 1; d < 16; d <<= 1)
#pragma unroll
      for (int r = 0; r < 8; r++)
        rmax[r] = fmaxf(rmax[r], __shfl_xor(rmax[r], d, 32));

    float alpha[8];
#pragma unroll
    for (int r = 0; r < 8; r++) {
      float mn = fmaxf(mrow[r], rmax[r]);
      alpha[r] = __expf(mrow[r] - mn);
      mrow[r] = mn;
    }

    // P = exp(S - m), masked keys -> 0; write bf16 P tile to LDS.
    float rsum[8] = {0.f, 0.f, 0.f, 0.f, 0.f, 0.f, 0.f, 0.f};
#pragma unroll
    for (int nt = 0; nt < 4; nt++)
#pragma unroll
      for (int r = 0; r < 8; r++) {
        float p = (keep[nt] != 0.f) ? __expf(s4[nt][r] - mrow[r]) : 0.f;
        rsum[r] += p;
        Ps[wave][r + 8 * hl][nt * 16 + ml] = (__bf16)p;
      }
#pragma unroll
    for (int d = 1; d < 16; d <<= 1)
#pragma unroll
      for (int r = 0; r < 8; r++) rsum[r] += __shfl_xor(rsum[r], d, 32);
#pragma unroll
    for (int r = 0; r < 8; r++) lrow[r] = lrow[r] * alpha[r] + rsum[r];
#pragma unroll
    for (int d = 0; d < 4; d++)
#pragma unroll
      for (int r = 0; r < 8; r++) o[d][r] *= alpha[r];

    __syncthreads();  // P tile visible (and ordered) before fragment reads

    // O += P * V.
    v16bf pf[2];
    pf[0] = fragA(&Ps[wave][ml][0], hl);
    pf[1] = fragA(&Ps[wave][ml][32], hl);
#pragma unroll
    for (int dt = 0; dt < 4; dt++) {
#pragma unroll
      for (int ks = 0; ks < 2; ks++) {
        v16bf vf = fragB(&Vs[dt * 16 + ml][32 * ks], hl);
        o[dt] = WMMA_BF16(pf[ks], vf, o[dt]);
      }
    }
  }

  // Normalize and store to [B][L][D] bf16 (heads concatenated).
  float inv[8];
#pragma unroll
  for (int r = 0; r < 8; r++) inv[r] = (lrow[r] > 0.f) ? (1.f / lrow[r]) : 0.f;
#pragma unroll
  for (int dt = 0; dt < 4; dt++)
#pragma unroll
    for (int r = 0; r < 8; r++) {
      const int gq = q0 + r + 8 * hl;
      const int col = h * Dh + dt * 16 + ml;
      Out[((size_t)b * Lseq + gq) * Dm + col] = (__bf16)(o[dt][r] * inv[r]);
    }
}

// ---------------------------------------------------------------------------
extern "C" void kernel_launch(void* const* d_in, const int* in_sizes, int n_in,
                              void* d_out, int out_size, void* d_ws, size_t ws_size,
                              hipStream_t stream) {
  (void)in_sizes; (void)n_in; (void)out_size; (void)ws_size;
  const float* q_x = (const float*)d_in[0];
  const float* k_x = (const float*)d_in[1];
  const float* v_x = (const float*)d_in[2];
  const float* ec  = (const float*)d_in[3];
  const float* Wq  = (const float*)d_in[4];
  const float* bq  = (const float*)d_in[5];
  const float* Wk  = (const float*)d_in[6];
  const float* bk  = (const float*)d_in[7];
  const float* Wv  = (const float*)d_in[8];
  const float* bv  = (const float*)d_in[9];
  const float* Wo  = (const float*)d_in[10];
  const float* bo  = (const float*)d_in[11];

  // Workspace: 4 x (8192*1024) bf16 arrays = 64 MB.
  __bf16* Qp = (__bf16*)d_ws;
  __bf16* Kp = Qp + (size_t)Mrows * Dm;
  __bf16* Vt = Kp + (size_t)Mrows * Dm;
  __bf16* AO = Vt + (size_t)Mrows * Dm;

  dim3 gg(Mrows / BM, Dm / BN);  // (64, 16)
  gemm_kernel<true, EP_Q><<<gg, 256, 0, stream>>>(q_x, Wq, bq, Qp);
  gemm_kernel<true, EP_K><<<gg, 256, 0, stream>>>(k_x, Wk, bk, Kp);
  gemm_kernel<true, EP_V><<<gg, 256, 0, stream>>>(v_x, Wv, bv, Vt);
  attn_kernel<<<Bb * Hh * (Lseq / 64), 128, 0, stream>>>(Qp, Kp, Vt, ec, AO);
  gemm_kernel<false, EP_OUT><<<gg, 256, 0, stream>>>(AO, Wo, bo, d_out);
}
